// DyMRconv_36258113913048
// MI455X (gfx1250) — matrix operand order repaired and checked by reading, not compiled, not run on hardware
//
#include <hip/hip_runtime.h>
#include <hip/hip_bf16.h>
#include <math.h>

// ---------------------------------------------------------------------------
// Problem constants (from reference: B=16, C=128, H=W=56, in=384, out=384)
// ---------------------------------------------------------------------------
#define B_    16
#define C_    128
#define H_    56
#define W_    56
#define HW_   (H_ * W_)          // 3136
#define KDIM  384                // 3*C (GEMM K)
#define KPAIRS (KDIM / 2)        // 192 packed bf16 pairs along K
#define MDIM  384                // out channels (GEMM M)
#define KSTRIDE 4
#define NSHIFT 14                // unique shifts 0,4,...,52 (each appears twice)
#define BN_EPS 1e-5f
#define KSTEPS (KDIM / 32)       // 12

typedef __attribute__((ext_vector_type(16))) __bf16 v16bf;
typedef __attribute__((ext_vector_type(8)))  float  v8f;

static __device__ __forceinline__ uint16_t f2bf(float f) {
    // round-to-nearest-even fp32 -> bf16
    uint32_t u = __float_as_uint(f);
    uint32_t r = (u + 0x7FFFu + ((u >> 16) & 1u)) >> 16;
    return (uint16_t)r;
}
static __device__ __forceinline__ uint32_t pack2bf(float f0, float f1) {
    return (uint32_t)f2bf(f0) | ((uint32_t)f2bf(f1) << 16);
}

// Async global->LDS copy (CDNA5, ASYNCcnt-tracked). LDS operand is the wave-
// relative LDS byte address = low 32 bits of the flat shared-aperture pointer
// (ISA 10.2).
static __device__ __forceinline__ void async_copy_b128(void* lds_ptr, const void* gptr) {
    uint32_t lds_off = (uint32_t)(uintptr_t)lds_ptr;
    asm volatile("global_load_async_to_lds_b128 %0, %1, off"
                 :: "v"(lds_off), "v"(gptr) : "memory");
}
static __device__ __forceinline__ void wait_asynccnt0() {
    asm volatile("s_wait_asynccnt 0" ::: "memory");
}

// ---------------------------------------------------------------------------
// 1) norm[b,h,w] = sum_c |x - roll(x, (H/2,W/2))|
// ---------------------------------------------------------------------------
__global__ void k_norm(const float* __restrict__ x, float* __restrict__ norm) {
    int idx = blockIdx.x * blockDim.x + threadIdx.x;
    if (idx >= B_ * HW_) return;
    int b = idx / HW_, hw = idx % HW_;
    int h = hw / W_, w = hw % W_;
    int h2 = (h + H_ / 2) % H_, w2 = (w + W_ / 2) % W_;
    const float* xb = x + (size_t)b * C_ * HW_;
    float s = 0.f;
    for (int c = 0; c < C_; ++c) {
        float a = xb[(size_t)c * HW_ + hw];
        float r = xb[(size_t)c * HW_ + h2 * W_ + w2];
        s += fabsf(a - r);
    }
    norm[idx] = s;
}

// ---------------------------------------------------------------------------
// 2) thresh[b] = mean(norm_b) - std(norm_b, ddof=1)
// ---------------------------------------------------------------------------
__global__ void k_stats(const float* __restrict__ norm, float* __restrict__ thresh) {
    __shared__ float s_sum[256], s_sq[256];
    int b = blockIdx.x;
    int tid = threadIdx.x;
    float sum = 0.f, sq = 0.f;
    for (int i = tid; i < HW_; i += 256) {
        float v = norm[b * HW_ + i];
        sum += v; sq += v * v;
    }
    s_sum[tid] = sum; s_sq[tid] = sq;
    __syncthreads();
    for (int off = 128; off > 0; off >>= 1) {
        if (tid < off) { s_sum[tid] += s_sum[tid + off]; s_sq[tid] += s_sq[tid + off]; }
        __syncthreads();
    }
    if (tid == 0) {
        float n = (float)HW_;
        float mean = s_sum[0] / n;
        float var  = (s_sq[0] - n * mean * mean) / (n - 1.0f);
        thresh[b] = mean - sqrtf(fmaxf(var, 0.f));
    }
}

// ---------------------------------------------------------------------------
// 3) shift loop: per (b,w) column in LDS; writes mask_sum + packed bf16
//    featP pairs for parts 0 (x) and 1 (x_j).
//    (shift list duplicated in reference -> mask counted twice)
// ---------------------------------------------------------------------------
__global__ void __launch_bounds__(256)
k_shift(const float* __restrict__ x, const float* __restrict__ thresh,
        uint32_t* __restrict__ featP, float* __restrict__ mask_sum) {
    __shared__ float col[C_][H_];      // 28 KB
    __shared__ float dist[NSHIFT][H_]; // ~3 KB
    int b = blockIdx.x / W_;
    int w = blockIdx.x % W_;
    const float* xb = x + (size_t)b * C_ * HW_ + w;
    for (int i = threadIdx.x; i < C_ * H_; i += 256) {
        int c = i / H_, h = i % H_;
        col[c][h] = xb[(size_t)(c * H_ + h) * W_];
    }
    __syncthreads();
    for (int t = threadIdx.x; t < NSHIFT * H_; t += 256) {
        int s = t / H_, h = t % H_;
        int hs = (h - s * KSTRIDE + H_) % H_;   // jnp.roll(x, s, axis=2)
        float d = 0.f;
        for (int c = 0; c < C_; ++c) d += fabsf(col[c][h] - col[c][hs]);
        dist[s][h] = d;
    }
    __syncthreads();
    float th = thresh[b];
    for (int h = threadIdx.x; h < H_; h += 256) {
        float ms = 0.f;
        for (int s = 0; s < NSHIFT; ++s) ms += (dist[s][h] < th) ? 2.0f : 0.0f;
        mask_sum[b * HW_ + h * W_ + w] = ms;
    }
    // channel pairs -> packed uint32 (k-pair layout used by the WMMA GEMM)
    for (int i = threadIdx.x; i < (C_ / 2) * H_; i += 256) {
        int cp = i / H_;            // channel pair 0..63
        int h  = i % H_;
        int c0 = 2 * cp;
        float xv0 = col[c0][h],     xv1 = col[c0 + 1][h];
        float xj0 = 0.f, xj1 = 0.f;
        #pragma unroll
        for (int s = 0; s < NSHIFT; ++s) {
            if (dist[s][h] < th) {
                int hs = (h - s * KSTRIDE + H_) % H_;
                xj0 = fmaxf(xj0, col[c0][hs]     - xv0);
                xj1 = fmaxf(xj1, col[c0 + 1][hs] - xv1);
            }
        }
        size_t hw = (size_t)h * W_ + w;
        featP[((size_t)b * KPAIRS + cp)            * HW_ + hw] = pack2bf(xv0, xv1);
        featP[((size_t)b * KPAIRS + C_ / 2 + cp)   * HW_ + hw] = pack2bf(xj0, xj1);
    }
}

// ---------------------------------------------------------------------------
// 4) thresh2[b] = 0.4 * (max - min) of mask_sum over batch
// ---------------------------------------------------------------------------
__global__ void k_minmax(const float* __restrict__ mask_sum, float* __restrict__ thresh2) {
    __shared__ float smin[256], smax[256];
    int b = blockIdx.x, tid = threadIdx.x;
    float mn = 3.0e38f, mx = -3.0e38f;
    for (int i = tid; i < HW_; i += 256) {
        float v = mask_sum[b * HW_ + i];
        mn = fminf(mn, v); mx = fmaxf(mx, v);
    }
    smin[tid] = mn; smax[tid] = mx;
    __syncthreads();
    for (int off = 128; off > 0; off >>= 1) {
        if (tid < off) {
            smin[tid] = fminf(smin[tid], smin[tid + off]);
            smax[tid] = fmaxf(smax[tid], smax[tid + off]);
        }
        __syncthreads();
    }
    if (tid == 0) thresh2[b] = (smax[0] - smin[0]) * 0.4f;
}

// ---------------------------------------------------------------------------
// 5) feat part 2 = x * (mask_sum < thresh2), packed channel pairs
// ---------------------------------------------------------------------------
__global__ void k_feat2(const float* __restrict__ x, const float* __restrict__ mask_sum,
                        const float* __restrict__ thresh2, uint32_t* __restrict__ featP) {
    int idx = blockIdx.x * blockDim.x + threadIdx.x;
    if (idx >= B_ * (C_ / 2) * HW_) return;
    int b = idx / ((C_ / 2) * HW_);
    int rem = idx % ((C_ / 2) * HW_);
    int cp = rem / HW_, hw = rem % HW_;
    float wgt = (mask_sum[b * HW_ + hw] < thresh2[b]) ? 1.0f : 0.0f;
    float x0 = x[((size_t)b * C_ + 2 * cp)     * HW_ + hw];
    float x1 = x[((size_t)b * C_ + 2 * cp + 1) * HW_ + hw];
    featP[((size_t)b * KPAIRS + C_ + cp) * HW_ + hw] = pack2bf(x0 * wgt, x1 * wgt);
}

// ---------------------------------------------------------------------------
// 6) one-time weight conversion: conv_w fp32 -> k-pair packed bf16 uint32
// ---------------------------------------------------------------------------
__global__ void k_packw(const float* __restrict__ conv_w, uint32_t* __restrict__ wP) {
    int idx = blockIdx.x * blockDim.x + threadIdx.x;
    if (idx >= MDIM * KPAIRS) return;
    int m = idx / KPAIRS, p = idx % KPAIRS;
    wP[idx] = pack2bf(conv_w[(size_t)m * KDIM + 2 * p],
                      conv_w[(size_t)m * KDIM + 2 * p + 1]);
}

// ---------------------------------------------------------------------------
// 7) zero BN accumulators
// ---------------------------------------------------------------------------
__global__ void k_zero(float* __restrict__ p, int n) {
    int i = blockIdx.x * blockDim.x + threadIdx.x;
    if (i < n) p[i] = 0.f;
}

// ---------------------------------------------------------------------------
// 8) WMMA GEMM: y[b,o,hw] = conv_w[o,:] . feat[b,:,hw] + bias
//    128(M)x64(N) per workgroup, 8 waves, each wave 16x64, K-step = 32 bf16.
//      A tile: async global->LDS copies (ASYNCcnt), no VGPR staging
//      B tile: 1x global b128 -> 4x ds_store_b32 (k-pair transpose)
//    Double-buffered LDS, k+2 prefetch; per-channel BN sums via atomics.
// ---------------------------------------------------------------------------
__global__ void __launch_bounds__(256)
k_gemm(const uint32_t* __restrict__ featP, const uint32_t* __restrict__ wP,
       const float* __restrict__ conv_b, float* __restrict__ y,
       float* __restrict__ bn_sum, float* __restrict__ bn_sq) {
    // LDS tiles: one uint32 = 2 consecutive K values (bf16 pair)
    __shared__ uint32_t Au[2][128 * 16]; // A: [mrow][kpair], 128x32 bf16 (8 KB each)
    __shared__ uint32_t Bu[2][64 * 16];  // B^T: [ncol][kpair], 64x32 bf16 (4 KB each)

    int b  = blockIdx.z;
    int m0 = blockIdx.y * 128;
    int n0 = blockIdx.x * 64;
    int tid  = threadIdx.x;
    int wv   = tid >> 5;       // wave id 0..7
    int lane = tid & 31;
    int half = lane >> 4;      // 0: lanes 0-15, 1: lanes 16-31
    int l16  = lane & 15;

    // per-thread load coordinates
    int arow = tid >> 1;            // A: row 0..127 (2 threads per row)
    int apo  = (tid & 1) * 8;       // A: kpair offset 0 or 8
    int bp   = tid >> 4;            // B: kpair 0..15 within the tile
    int bc4  = (tid & 15) * 4;      // B: 4 hw-columns

    const uint32_t* wp_base = wP + (size_t)(m0 + arow) * KPAIRS + apo;
    const uint32_t* fp_base = featP + ((size_t)b * KPAIRS + bp) * HW_ + n0 + bc4;

    v8f acc[4] = {};

    // --- prologue: tile 0 -> buffer 0
    {
        async_copy_b128(&Au[0][arow * 16 + apo],     wp_base);
        async_copy_b128(&Au[0][arow * 16 + apo + 4], wp_base + 4);
        uint4 bb = *(const uint4*)(fp_base);
        Bu[0][(bc4 + 0) * 16 + bp] = bb.x;
        Bu[0][(bc4 + 1) * 16 + bp] = bb.y;
        Bu[0][(bc4 + 2) * 16 + bp] = bb.z;
        Bu[0][(bc4 + 3) * 16 + bp] = bb.w;
        wait_asynccnt0();
    }

    for (int kt = 0; kt < KSTEPS; ++kt) {
        int buf = kt & 1;
        __syncthreads();   // tile[buf] ready; prior compute on buf^1 done

        // --- fragment loads from LDS (distinct registers for all fragments)
        // A (ISA 16-bit A 16x32 layout): lane m=l16, VGPR v holds
        //   K = (v/4)*16 + half*8 + (v%4)*2, +1
        union { v16bf v; uint32_t u[8]; } afr;
        #pragma unroll
        for (int v = 0; v < 8; ++v) {
            int kp = (v >> 2) * 8 + half * 4 + (v & 3);
            afr.u[v] = Au[buf][(wv * 16 + l16) * 16 + kp];
        }
        union { v16bf v; uint32_t u[8]; } bfr[4];
        #pragma unroll
        for (int s = 0; s < 4; ++s) {
            int n = s * 16 + l16;     // N = l16; same K layout as A
            #pragma unroll
            for (int v = 0; v < 8; ++v)
                bfr[s].u[v] = Bu[buf][n * 16 + (v >> 2) * 8 + half * 4 + (v & 3)];
        }

        // --- kick off next tile: A via async copies, B staged in registers
        bool more = (kt + 1 < KSTEPS);
        int nb = buf ^ 1;
        uint4 bb;
        if (more) {
            async_copy_b128(&Au[nb][arow * 16 + apo],     wp_base + (kt + 1) * 16);
            async_copy_b128(&Au[nb][arow * 16 + apo + 4], wp_base + (kt + 1) * 16 + 4);
            bb = *(const uint4*)(fp_base + (size_t)((kt + 1) * 16) * HW_);
            if (kt + 2 < KSTEPS) {
                __builtin_prefetch(fp_base + (size_t)((kt + 2) * 16) * HW_, 0, 1);
                __builtin_prefetch(wp_base + (kt + 2) * 16, 0, 1);
            }
        }

        // --- 4 back-to-back WMMAs (independent accumulators)
        acc[0] = __builtin_amdgcn_wmma_f32_16x16x32_bf16(
            false, afr.v, false, bfr[0].v, (short)0, acc[0], false, false);
        acc[1] = __builtin_amdgcn_wmma_f32_16x16x32_bf16(
            false, afr.v, false, bfr[1].v, (short)0, acc[1], false, false);
        acc[2] = __builtin_amdgcn_wmma_f32_16x16x32_bf16(
            false, afr.v, false, bfr[2].v, (short)0, acc[2], false, false);
        acc[3] = __builtin_amdgcn_wmma_f32_16x16x32_bf16(
            false, afr.v, false, bfr[3].v, (short)0, acc[3], false, false);

        // --- write staged B tile, then wait for async A before the barrier
        if (more) {
            Bu[nb][(bc4 + 0) * 16 + bp] = bb.x;
            Bu[nb][(bc4 + 1) * 16 + bp] = bb.y;
            Bu[nb][(bc4 + 2) * 16 + bp] = bb.z;
            Bu[nb][(bc4 + 3) * 16 + bp] = bb.w;
            wait_asynccnt0();
        }
    }

    // --- epilogue: bias, store y, BN partial stats
    float* yb = y + (size_t)b * MDIM * HW_;
    #pragma unroll
    for (int r = 0; r < 8; ++r) {
        int mabs = m0 + wv * 16 + r + 8 * half;   // C/D layout: M = r + 8*half
        float bias = conv_b[mabs];
        float sum = 0.f, sq = 0.f;
        #pragma unroll
        for (int s = 0; s < 4; ++s) {
            float v = acc[s][r] + bias;
            yb[(size_t)mabs * HW_ + n0 + s * 16 + l16] = v;
            sum += v; sq += v * v;
        }
        // butterfly over the 16 lanes sharing this M (xor masks stay in half-group)
        #pragma unroll
        for (int off = 1; off < 16; off <<= 1) {
            sum += __shfl_xor(sum, off, 32);
            sq  += __shfl_xor(sq,  off, 32);
        }
        if (l16 == 0) {
            atomicAdd(&bn_sum[mabs], sum);
            atomicAdd(&bn_sq[mabs],  sq);
        }
    }
}

// ---------------------------------------------------------------------------
// 9) BatchNorm (batch stats, biased var) + gamma/beta + exact GELU
// ---------------------------------------------------------------------------
__global__ void k_bn_gelu(const float* __restrict__ y, const float* __restrict__ bn_sum,
                          const float* __restrict__ bn_sq, const float* __restrict__ gamma,
                          const float* __restrict__ beta, float* __restrict__ out) {
    size_t idx = (size_t)blockIdx.x * blockDim.x + threadIdx.x;
    size_t total = (size_t)B_ * MDIM * HW_;
    if (idx >= total) return;
    int o = (int)((idx / HW_) % MDIM);
    float n = (float)(B_ * HW_);
    float mean = bn_sum[o] / n;
    float var  = bn_sq[o] / n - mean * mean;
    float v = (y[idx] - mean) * rsqrtf(var + BN_EPS);
    v = v * gamma[o] + beta[o];
    out[idx] = 0.5f * v * (1.0f + erff(v * 0.70710678118654752f));
}

// ---------------------------------------------------------------------------
// launch
// ---------------------------------------------------------------------------
extern "C" void kernel_launch(void* const* d_in, const int* in_sizes, int n_in,
                              void* d_out, int out_size, void* d_ws, size_t ws_size,
                              hipStream_t stream) {
    (void)in_sizes; (void)n_in; (void)out_size; (void)ws_size;
    const float* x      = (const float*)d_in[0];
    const float* conv_w = (const float*)d_in[1];
    const float* conv_b = (const float*)d_in[2];
    const float* gamma  = (const float*)d_in[3];
    const float* beta   = (const float*)d_in[4];
    float* out = (float*)d_out;

    char* ws = (char*)d_ws;
    size_t off = 0;
    auto carve = [&](size_t bytes) -> void* {
        void* p = ws + off;
        off += (bytes + 255) & ~(size_t)255;
        return p;
    };
    uint32_t* featP  = (uint32_t*)carve((size_t)B_ * KPAIRS * HW_ * sizeof(uint32_t));
    float*    yb     = (float*)   carve((size_t)B_ * MDIM * HW_ * sizeof(float));
    uint32_t* wP     = (uint32_t*)carve((size_t)MDIM * KPAIRS * sizeof(uint32_t));
    float*    norm   = (float*)   carve((size_t)B_ * HW_ * sizeof(float));
    float*    msum   = (float*)   carve((size_t)B_ * HW_ * sizeof(float));
    float*    thresh = (float*)   carve(64);
    float*    th2    = (float*)   carve(64);
    float*    bn     = (float*)   carve(2 * MDIM * sizeof(float));
    float*    bn_sum = bn;
    float*    bn_sq  = bn + MDIM;

    k_norm<<<(B_ * HW_ + 255) / 256, 256, 0, stream>>>(x, norm);
    k_stats<<<B_, 256, 0, stream>>>(norm, thresh);
    k_shift<<<B_ * W_, 256, 0, stream>>>(x, thresh, featP, msum);
    k_minmax<<<B_, 256, 0, stream>>>(msum, th2);
    k_feat2<<<(B_ * (C_ / 2) * HW_ + 255) / 256, 256, 0, stream>>>(x, msum, th2, featP);
    k_packw<<<(MDIM * KPAIRS + 255) / 256, 256, 0, stream>>>(conv_w, wP);
    k_zero<<<(2 * MDIM + 255) / 256, 256, 0, stream>>>(bn, 2 * MDIM);
    k_gemm<<<dim3(HW_ / 64, MDIM / 128, B_), 256, 0, stream>>>(featP, wP, conv_b,
                                                               yb, bn_sum, bn_sq);
    k_bn_gelu<<<(int)(((size_t)B_ * MDIM * HW_ + 255) / 256), 256, 0, stream>>>(
        yb, bn_sum, bn_sq, gamma, beta, out);
}
